// Decoder_23141283791209
// MI455X (gfx1250) — matrix operand order repaired and checked by reading, not compiled, not run on hardware
//
#include <hip/hip_runtime.h>
#include <cstdint>

#define DEVINL __device__ __forceinline__

typedef __attribute__((ext_vector_type(16))) __bf16 v16bf;
typedef __attribute__((ext_vector_type(8)))  float  v8f;
typedef unsigned short u16;

namespace {
constexpr int B_ = 4, T_ = 1024, E_ = 1024, H_ = 16, F_ = 64;
constexpr size_t TE  = (size_t)T_ * E_;
constexpr size_t NTE = (size_t)B_ * TE;
constexpr int NPART = 512;                 // per-batch partial-reduction blocks

constexpr size_t MB = 1ull << 20;
constexpr size_t OFF_XB  = 0;              // bf16 x        [B,T,E]    8MB
constexpr size_t OFF_QB  = 8  * MB;        // bf16 q        [B,H,T,F]  8MB
constexpr size_t OFF_KB  = 16 * MB;        // bf16 k        [B,H,T,F]  8MB
constexpr size_t OFF_VT  = 24 * MB;        // bf16 v^T      [B,H,F,T]  8MB
constexpr size_t OFF_WQT = 32 * MB;        // bf16 q_w^T    [H,F,E]    2MB
constexpr size_t OFF_WKT = 34 * MB;
constexpr size_t OFF_WVT = 36 * MB;
constexpr size_t OFF_FRT = 38 * MB;        // bf16 fr_w^T   [B,E,E]    8MB
constexpr size_t OFF_FFB = 46 * MB;        // bf16 ff_w     [E,E]      2MB
constexpr size_t OFF_ZB  = 48 * MB;        // bf16 attn-out [B,T,E]    8MB
constexpr size_t OFF_S1  = 56 * MB;        // f32 residual  [B,T,E]   16MB (reused as s2)
constexpr size_t OFF_Z1  = 72 * MB;        // f32 z1        [B,T,E]   16MB
constexpr size_t OFF_Z1B = 88 * MB;        // bf16 z1       [B,T,E]    8MB
constexpr size_t OFF_RED = 96 * MB;        // partial sums  (B*NPART*2 f32)
constexpr size_t OFF_STA = 96 * MB + 64 * 1024;  // mean/invstd per batch
} // namespace

DEVINL u16 f2bf(float f) {                 // round-to-nearest-even f32 -> bf16
  unsigned u = __float_as_uint(f);
  unsigned r = u + 0x7FFFu + ((u >> 16) & 1u);
  return (u16)(r >> 16);
}

DEVINL v16bf mk_frag(const u16* p0, const u16* p1) {
  union { v16bf v; uint4 u[2]; } f;
  f.u[0] = *(const uint4*)p0;
  f.u[1] = *(const uint4*)p1;
  return f.v;
}

DEVINL v8f wmma_bf16(v16bf a, v16bf b, v8f c) {
  return __builtin_amdgcn_wmma_f32_16x16x32_bf16(false, a, false, b, (short)0, c,
                                                 false, false);
}

// ---------------- conversion kernels ----------------
__global__ void k_cvt(const float* __restrict__ src, u16* __restrict__ dst, size_t n4) {
  size_t i = (size_t)blockIdx.x * blockDim.x + threadIdx.x;
  if (i >= n4) return;
  float4 f = ((const float4*)src)[i];
  ushort4 o;
  o.x = f2bf(f.x); o.y = f2bf(f.y); o.z = f2bf(f.z); o.w = f2bf(f.w);
  ((ushort4*)dst)[i] = o;
}

__global__ void k_cvt_wT(const float* __restrict__ w, u16* __restrict__ wt) {
  // [H,E,F] -> [H,F,E] bf16
  size_t i = (size_t)blockIdx.x * blockDim.x + threadIdx.x;   // < H*E*F
  int h = (int)(i / (size_t)(E_ * F_));
  int rem = (int)(i % (size_t)(E_ * F_));
  int e = rem / F_, f = rem % F_;
  wt[((size_t)h * F_ + f) * E_ + e] = f2bf(w[i]);
}

__global__ void k_cvt_frT(const float* __restrict__ w, u16* __restrict__ wt) {
  // [B,E(src),E(dst)] -> [B,E(dst),E(src)] bf16
  size_t i = (size_t)blockIdx.x * blockDim.x + threadIdx.x;   // < B*E*E
  size_t b = i / ((size_t)E_ * E_);
  size_t rem = i % ((size_t)E_ * E_);
  int e = (int)(rem / E_), d = (int)(rem % E_);
  wt[((size_t)b * E_ + d) * E_ + e] = f2bf(w[i]);
}

// ---------------- QKV projection (one matrix per blockIdx.y, pipelined) ----------------
__global__ void k_qkv(const u16* __restrict__ xb, const u16* __restrict__ wqt,
                      const u16* __restrict__ wkt, const u16* __restrict__ wvt,
                      u16* __restrict__ qb, u16* __restrict__ kbuf,
                      u16* __restrict__ vt) {
  const int wave = threadIdx.x >> 5, lane = threadIdx.x & 31;
  const int mt  = blockIdx.x;          // T/128 row tiles
  const int mat = blockIdx.y;          // 0=q, 1=k, 2=v
  const int bh  = blockIdx.z;          // b*H + h
  const int h   = bh & 15;
  const int r0  = mt * 128 + wave * 16;
  const int nn  = lane & 15, hi = lane >> 4;
  const int kbA = 8 * hi, kbB = 16 * hi;

  const int b = bh >> 4;
  const u16* A = xb + (size_t)b * TE + (size_t)(r0 + nn) * E_;
  const u16* W = ((mat == 0) ? wqt : (mat == 1) ? wkt : wvt) + (size_t)h * F_ * E_;

  v8f acc[4] = {};
  // register double-buffer: prefetch next K-step while WMMAs consume current
  v16bf afc = mk_frag(A + kbB, A + kbB + 8);
  v16bf bfc[4];
#pragma unroll
  for (int j = 0; j < 4; ++j) {
    const u16* bp = W + (size_t)(j * 16 + nn) * E_ + kbB;
    bfc[j] = mk_frag(bp, bp + 8);
  }
  for (int k0 = 0; k0 < E_; k0 += 32) {
    const int kn = (k0 + 32 < E_) ? (k0 + 32) : 0;   // last iter: harmless re-load
    v16bf afn = mk_frag(A + kn + kbB, A + kn + kbB + 8);
    v16bf bfn[4];
#pragma unroll
    for (int j = 0; j < 4; ++j) {
      const u16* bp = W + (size_t)(j * 16 + nn) * E_ + kn + kbB;
      bfn[j] = mk_frag(bp, bp + 8);
    }
#pragma unroll
    for (int j = 0; j < 4; ++j) acc[j] = wmma_bf16(afc, bfc[j], acc[j]);
    afc = afn;
#pragma unroll
    for (int j = 0; j < 4; ++j) bfc[j] = bfn[j];
  }

  const size_t obase = (size_t)bh * T_;
  if (mat == 2) {            // V stored transposed [B,H,F,T]
#pragma unroll
    for (int j = 0; j < 4; ++j)
#pragma unroll
      for (int v = 0; v < 8; ++v) {
        const int t = r0 + v + 8 * hi, f = j * 16 + nn;
        vt[((size_t)bh * F_ + f) * T_ + t] = f2bf(acc[j][v]);
      }
  } else {
    u16* O = mat ? kbuf : qb;
#pragma unroll
    for (int j = 0; j < 4; ++j)
#pragma unroll
      for (int v = 0; v < 8; ++v) {
        const int t = r0 + v + 8 * hi, f = j * 16 + nn;
        O[(obase + t) * F_ + f] = f2bf(acc[j][v]);
      }
  }
}

// ---------------- flash attention (multiplicative mask, full row) ----------------
DEVINL void load_kfrags(const u16* K, int j0, int nn, int kbB, v16bf* kf) {
  const u16* kp0 = K + (size_t)(j0 + nn) * F_;
  const u16* kp1 = K + (size_t)(j0 + 16 + nn) * F_;
  kf[0] = mk_frag(kp0 + kbB,      kp0 + kbB + 8);
  kf[1] = mk_frag(kp0 + 32 + kbB, kp0 + 32 + kbB + 8);
  kf[2] = mk_frag(kp1 + kbB,      kp1 + kbB + 8);
  kf[3] = mk_frag(kp1 + 32 + kbB, kp1 + 32 + kbB + 8);
}

__global__ void k_attn(const u16* __restrict__ qb, const u16* __restrict__ kb,
                       const u16* __restrict__ vt, u16* __restrict__ zb) {
  __shared__ __align__(16) u16 ldsP[8][16][32];   // per-wave P tile bounce
  const int wave = threadIdx.x >> 5, lane = threadIdx.x & 31;
  const int mt = blockIdx.x & 7;
  const int bh = blockIdx.x >> 3;
  const int b  = bh >> 4, h = bh & 15;
  const int q0 = mt * 128 + wave * 16;
  const int nn = lane & 15, hi = lane >> 4;
  const int kbA = 8 * hi, kbB = 16 * hi;

  const u16* Q = qb + (size_t)bh * T_ * F_;
  const u16* K = kb + (size_t)bh * T_ * F_;
  const u16* V = vt + (size_t)bh * F_ * T_;

  const u16* qp = Q + (size_t)(q0 + nn) * F_;
  const v16bf aq0 = mk_frag(qp + kbB,      qp + kbB + 8);
  const v16bf aq1 = mk_frag(qp + 32 + kbB, qp + 32 + kbB + 8);

  v8f acc[4] = {};
  float ms[8], ls[8];
#pragma unroll
  for (int v = 0; v < 8; ++v) { ms[v] = -3.0e38f; ls[v] = 0.f; }

  v16bf kf[4];
  load_kfrags(K, 0, nn, kbB, kf);      // preload first key tile

  for (int j0 = 0; j0 < T_; j0 += 32) {
    // V fragments issued early — latency covered by S-WMMAs + softmax VALU
    v16bf vf[4];
#pragma unroll
    for (int j = 0; j < 4; ++j) {
      const u16* vp = V + (size_t)(j * 16 + nn) * T_ + j0 + kbB;
      vf[j] = mk_frag(vp, vp + 8);
    }

    v8f st0 = {}, st1 = {};
    st0 = wmma_bf16(aq0, kf[0], st0);
    st0 = wmma_bf16(aq1, kf[1], st0);
    st1 = wmma_bf16(aq0, kf[2], st1);
    st1 = wmma_bf16(aq1, kf[3], st1);

    // prefetch next key tile while softmax runs
    const int jn = (j0 + 32 < T_) ? (j0 + 32) : 0;
    v16bf kfn[4];
    load_kfrags(K, jn, nn, kbB, kfn);

#pragma unroll
    for (int v = 0; v < 8; ++v) {
      const int qg  = q0 + v + 8 * hi;
      const int k0i = j0 + nn, k1i = j0 + 16 + nn;
      // faithful multiplicative mask: strict-upper entries scaled by (-1e9 + 1)
      float a0 = st0[v] * ((k0i > qg) ? (-1.0e9f + 1.0f) : 1.0f) * 0.125f;
      float a1 = st1[v] * ((k1i > qg) ? (-1.0e9f + 1.0f) : 1.0f) * 0.125f;
      float mx = fmaxf(a0, a1);
      mx = fmaxf(mx, __shfl_xor(mx, 1, 32));
      mx = fmaxf(mx, __shfl_xor(mx, 2, 32));
      mx = fmaxf(mx, __shfl_xor(mx, 4, 32));
      mx = fmaxf(mx, __shfl_xor(mx, 8, 32));
      const float mn = fmaxf(ms[v], mx);
      const float sc = __expf(ms[v] - mn);
      ms[v] = mn;
      const float p0 = __expf(a0 - mn), p1 = __expf(a1 - mn);
      float rs = p0 + p1;
      rs += __shfl_xor(rs, 1, 32);
      rs += __shfl_xor(rs, 2, 32);
      rs += __shfl_xor(rs, 4, 32);
      rs += __shfl_xor(rs, 8, 32);
      ls[v] = ls[v] * sc + rs;
#pragma unroll
      for (int j = 0; j < 4; ++j) acc[j][v] *= sc;
      const int m = v + 8 * hi;
      ldsP[wave][m][nn]      = f2bf(p0);
      ldsP[wave][m][nn + 16] = f2bf(p1);
    }
    // same-wave LDS ops are in-order; one dscnt wait makes the C->A relayout safe
    asm volatile("s_wait_dscnt 0x0" ::: "memory");
    const v16bf pf = mk_frag(&ldsP[wave][nn][kbB], &ldsP[wave][nn][kbB + 8]);
#pragma unroll
    for (int j = 0; j < 4; ++j) acc[j] = wmma_bf16(pf, vf[j], acc[j]);
#pragma unroll
    for (int j = 0; j < 4; ++j) kf[j] = kfn[j];
  }
#pragma unroll
  for (int j = 0; j < 4; ++j)
#pragma unroll
    for (int v = 0; v < 8; ++v) {
      const int t = q0 + v + 8 * hi, f = j * 16 + nn;
      zb[((size_t)b * T_ + t) * E_ + h * F_ + f] = f2bf(acc[j][v] / ls[v]);
    }
}

// ---------------- generic 1024x1024 GEMM + epilogue (pipelined) ----------------
__global__ void k_gemm_ep(const u16* __restrict__ A, const u16* __restrict__ Bt,
                          const float* __restrict__ resid, const float* __restrict__ bias,
                          float* __restrict__ out, size_t bBatch, int relu) {
  const int wave = threadIdx.x >> 5, lane = threadIdx.x & 31;
  const int bz = blockIdx.z;
  const int r0 = blockIdx.x * 128 + wave * 16;
  const int c0 = blockIdx.y * 64;
  const int nn = lane & 15, hi = lane >> 4;
  const int kbA = 8 * hi, kbB = 16 * hi;

  const u16*   Ab = A     + (size_t)bz * TE + (size_t)(r0 + nn) * E_;
  const u16*   Bb = Bt    + (size_t)bz * bBatch;
  const float* Rb = resid + (size_t)bz * TE;
  float*       Ob = out   + (size_t)bz * TE;

  v8f acc[4] = {};
  v16bf afc = mk_frag(Ab + kbB, Ab + kbB + 8);
  v16bf bfc[4];
#pragma unroll
  for (int j = 0; j < 4; ++j) {
    const u16* bp = Bb + (size_t)(c0 + j * 16 + nn) * E_ + kbB;
    bfc[j] = mk_frag(bp, bp + 8);
  }
  for (int k0 = 0; k0 < E_; k0 += 32) {
    const int kn = (k0 + 32 < E_) ? (k0 + 32) : 0;
    v16bf afn = mk_frag(Ab + kn + kbB, Ab + kn + kbB + 8);
    v16bf bfn[4];
#pragma unroll
    for (int j = 0; j < 4; ++j) {
      const u16* bp = Bb + (size_t)(c0 + j * 16 + nn) * E_ + kn + kbB;
      bfn[j] = mk_frag(bp, bp + 8);
    }
#pragma unroll
    for (int j = 0; j < 4; ++j) acc[j] = wmma_bf16(afc, bfc[j], acc[j]);
    afc = afn;
#pragma unroll
    for (int j = 0; j < 4; ++j) bfc[j] = bfn[j];
  }
#pragma unroll
  for (int j = 0; j < 4; ++j)
#pragma unroll
    for (int v = 0; v < 8; ++v) {
      const int r = r0 + v + 8 * hi, c = c0 + j * 16 + nn;
      float val = acc[j][v];
      if (bias) val += bias[c];
      if (relu) val = fmaxf(val, 0.f);
      Ob[(size_t)r * E_ + c] = Rb[(size_t)r * E_ + c] + val;
    }
}

// ---------------- deterministic [T,E] LayerNorm ----------------
__global__ void k_ln_part(const float* __restrict__ s, float* __restrict__ ps,
                          float* __restrict__ pq) {
  __shared__ float lsum[256], lsq[256];
  const int b = blockIdx.y;
  const float* p = s + (size_t)b * TE + (size_t)blockIdx.x * 2048 + threadIdx.x * 8;
  float4 a = ((const float4*)p)[0];
  float4 c = ((const float4*)p)[1];
  float su = a.x + a.y + a.z + a.w + c.x + c.y + c.z + c.w;
  float sq = a.x * a.x + a.y * a.y + a.z * a.z + a.w * a.w +
             c.x * c.x + c.y * c.y + c.z * c.z + c.w * c.w;
  lsum[threadIdx.x] = su; lsq[threadIdx.x] = sq;
  __syncthreads();
  for (int st = 128; st > 0; st >>= 1) {
    if ((int)threadIdx.x < st) {
      lsum[threadIdx.x] += lsum[threadIdx.x + st];
      lsq[threadIdx.x]  += lsq[threadIdx.x + st];
    }
    __syncthreads();
  }
  if (threadIdx.x == 0) {
    ps[b * NPART + blockIdx.x] = lsum[0];
    pq[b * NPART + blockIdx.x] = lsq[0];
  }
}

__global__ void k_ln_fin(const float* __restrict__ ps, const float* __restrict__ pq,
                         float* __restrict__ stats) {
  __shared__ float lsum[256], lsq[256];
  const int b = blockIdx.x;
  lsum[threadIdx.x] = ps[b * NPART + threadIdx.x] + ps[b * NPART + threadIdx.x + 256];
  lsq[threadIdx.x]  = pq[b * NPART + threadIdx.x] + pq[b * NPART + threadIdx.x + 256];
  __syncthreads();
  for (int st = 128; st > 0; st >>= 1) {
    if ((int)threadIdx.x < st) {
      lsum[threadIdx.x] += lsum[threadIdx.x + st];
      lsq[threadIdx.x]  += lsq[threadIdx.x + st];
    }
    __syncthreads();
  }
  if (threadIdx.x == 0) {
    const float mean = lsum[0] / (float)TE;
    const float var  = lsq[0] / (float)TE - mean * mean;
    stats[b * 2]     = mean;
    stats[b * 2 + 1] = rsqrtf(var + 1e-5f);
  }
}

__global__ void k_ln_apply(const float* __restrict__ s, const float* __restrict__ w,
                           const float* __restrict__ bw, const float* __restrict__ stats,
                           float* __restrict__ outF, u16* __restrict__ outB) {
  const size_t base = ((size_t)blockIdx.x * blockDim.x + threadIdx.x) * 4;
  const int b = (int)(base / TE);
  const size_t te = base % TE;
  const float mean = stats[b * 2], inv = stats[b * 2 + 1];
  float4 x  = ((const float4*)(s + base))[0];
  float4 wv = ((const float4*)(w + te))[0];
  float4 bv = ((const float4*)(bw + te))[0];
  float4 o;
  o.x = (x.x - mean) * inv * wv.x + bv.x;
  o.y = (x.y - mean) * inv * wv.y + bv.y;
  o.z = (x.z - mean) * inv * wv.z + bv.z;
  o.w = (x.w - mean) * inv * wv.w + bv.w;
  ((float4*)(outF + base))[0] = o;
  if (outB) {
    ushort4 ob;
    ob.x = f2bf(o.x); ob.y = f2bf(o.y); ob.z = f2bf(o.z); ob.w = f2bf(o.w);
    ((ushort4*)(outB + base))[0] = ob;
  }
}

// ---------------- host-side orchestration ----------------
extern "C" void kernel_launch(void* const* d_in, const int* in_sizes, int n_in,
                              void* d_out, int out_size, void* d_ws, size_t ws_size,
                              hipStream_t stream) {
  const float* x     = (const float*)d_in[0];
  const float* q_w   = (const float*)d_in[1];
  const float* k_w   = (const float*)d_in[2];
  const float* v_w   = (const float*)d_in[3];
  const float* fr_w  = (const float*)d_in[4];
  const float* ff_w  = (const float*)d_in[5];
  const float* ff_b  = (const float*)d_in[6];
  const float* ln1_w = (const float*)d_in[7];
  const float* ln1_b = (const float*)d_in[8];
  const float* ln2_w = (const float*)d_in[9];
  const float* ln2_b = (const float*)d_in[10];

  char* ws = (char*)d_ws;
  u16*   xb    = (u16*)(ws + OFF_XB);
  u16*   qb    = (u16*)(ws + OFF_QB);
  u16*   kbuf  = (u16*)(ws + OFF_KB);
  u16*   vtb   = (u16*)(ws + OFF_VT);
  u16*   wqt   = (u16*)(ws + OFF_WQT);
  u16*   wkt   = (u16*)(ws + OFF_WKT);
  u16*   wvt   = (u16*)(ws + OFF_WVT);
  u16*   frt   = (u16*)(ws + OFF_FRT);
  u16*   ffb   = (u16*)(ws + OFF_FFB);
  u16*   zb    = (u16*)(ws + OFF_ZB);
  float* s1    = (float*)(ws + OFF_S1);
  float* z1    = (float*)(ws + OFF_Z1);
  u16*   z1b   = (u16*)(ws + OFF_Z1B);
  float* psum  = (float*)(ws + OFF_RED);
  float* pq    = psum + B_ * NPART;
  float* stats = (float*)(ws + OFF_STA);

  // 1) precision / layout conversions
  k_cvt    <<<(int)(NTE / 1024), 256, 0, stream>>>(x, xb, NTE / 4);
  k_cvt    <<<(E_ * E_) / 1024,  256, 0, stream>>>(ff_w, ffb, (size_t)E_ * E_ / 4);
  k_cvt_wT <<<(H_ * E_ * F_) / 256, 256, 0, stream>>>(q_w, wqt);
  k_cvt_wT <<<(H_ * E_ * F_) / 256, 256, 0, stream>>>(k_w, wkt);
  k_cvt_wT <<<(H_ * E_ * F_) / 256, 256, 0, stream>>>(v_w, wvt);
  k_cvt_frT<<<(B_ * E_ * E_) / 256, 256, 0, stream>>>(fr_w, frt);

  // 2) QKV projection (y selects q/k/v), 3) flash attention
  dim3 gq(T_ / 128, 3, B_ * H_);
  k_qkv <<<gq, 256, 0, stream>>>(xb, wqt, wkt, wvt, qb, kbuf, vtb);
  dim3 g1(B_ * H_ * (T_ / 128));
  k_attn<<<g1, 256, 0, stream>>>(qb, kbuf, vtb, zb);

  // 4) per-batch feature reduction + residual(x) -> s1
  dim3 g2(T_ / 128, E_ / 64, B_);
  k_gemm_ep<<<g2, 256, 0, stream>>>(zb, frt, x, nullptr, s1, (size_t)E_ * E_, 0);

  // 5) LN1 -> z1 (f32 + bf16)
  dim3 gp(NPART, B_);
  k_ln_part <<<gp, 256, 0, stream>>>(s1, psum, pq);
  k_ln_fin  <<<B_, 256, 0, stream>>>(psum, pq, stats);
  k_ln_apply<<<(int)(NTE / 1024), 256, 0, stream>>>(s1, ln1_w, ln1_b, stats, z1, z1b);

  // 6) FFN (bias+ReLU) + residual(z1) -> s2 (reuses s1 buffer)
  k_gemm_ep<<<g2, 256, 0, stream>>>(z1b, ffb, z1, ff_b, s1, 0, 1);

  // 7) LN2 -> d_out
  k_ln_part <<<gp, 256, 0, stream>>>(s1, psum, pq);
  k_ln_fin  <<<B_, 256, 0, stream>>>(psum, pq, stats);
  k_ln_apply<<<(int)(NTE / 1024), 256, 0, stream>>>(s1, ln2_w, ln2_b, stats,
                                                    (float*)d_out, nullptr);
}
